// MultiHeadAttention_29111288332749
// MI455X (gfx1250) — matrix operand order repaired and checked by reading, not compiled, not run on hardware
//
#include <hip/hip_runtime.h>
#include <hip/hip_bf16.h>

#define D_MODEL 1024
#define N_HEADS 16
#define HEAD_DIM 64
#define SEQ 2048
#define BATCH 2
#define ROWS (BATCH * SEQ) // 4096

typedef __attribute__((ext_vector_type(16))) _Float16 v16h;
typedef __attribute__((ext_vector_type(8)))  _Float16 v8h;
typedef __attribute__((ext_vector_type(8)))  float    v8f;
typedef int v4i __attribute__((vector_size(16))); // matches builtin param type

union HF { v16h v; v8h h[2]; };

__device__ __forceinline__ v8f wmma16(v16h a, v16h b, v8f c) {
  // 8 args: (neg_a, A, neg_b, B, c_mod, C, reuse_a, reuse_b)
  return __builtin_amdgcn_wmma_f32_16x16x32_f16(false, a, false, b, (short)0, c,
                                                false, false);
}

// ---------------- async global->LDS staging (guarded) ----------------
#if defined(__has_builtin)
#  if __has_builtin(__builtin_amdgcn_global_load_async_to_lds_b128)
#    define USE_ASYNC_LDS 1
#  endif
#endif
#ifndef USE_ASYNC_LDS
#  define USE_ASYNC_LDS 0
#endif

__device__ __forceinline__ void copy16_g2l(void* lds, const void* glob) {
#if USE_ASYNC_LDS
  __builtin_amdgcn_global_load_async_to_lds_b128((v4i*)glob, (v4i*)lds, 0, 0);
#else
  *(v8h*)lds = *(const v8h*)glob;
#endif
}

__device__ __forceinline__ void async_fence() {
#if USE_ASYNC_LDS && __has_builtin(__builtin_amdgcn_s_wait_asynccnt)
  __builtin_amdgcn_s_wait_asynccnt(0);
#elif USE_ASYNC_LDS
  asm volatile("s_wait_asynccnt 0" ::: "memory");
#endif
}

// ---------------------------------------------------------------- conversions

__global__ void cvt_f32_to_f16(const float* __restrict__ s,
                               _Float16* __restrict__ d, int n) {
  int i = blockIdx.x * 256 + threadIdx.x;
  if (i < n) d[i] = (_Float16)s[i];
}

// dst[o*1024 + i] = (f16) src[i*1024 + o]   (W stored [in,out] -> Wt [out,in])
__global__ void transpose_cvt_w(const float* __restrict__ src,
                                _Float16* __restrict__ dst) {
  __shared__ __align__(16) _Float16 t[32][33];
  int o0 = blockIdx.x * 32, i0 = blockIdx.y * 32;
  int tx = threadIdx.x, ty = threadIdx.y;
#pragma unroll
  for (int j = 0; j < 4; ++j)
    t[ty + 8 * j][tx] =
        (_Float16)src[(size_t)(i0 + ty + 8 * j) * D_MODEL + o0 + tx];
  __syncthreads();
#pragma unroll
  for (int j = 0; j < 4; ++j)
    dst[(size_t)(o0 + ty + 8 * j) * D_MODEL + i0 + tx] = t[tx][ty + 8 * j];
}

// Vt[(bh*64 + d)*SEQ + s] = Vh[(b*SEQ + s)*1024 + h*64 + d]
__global__ void transpose_v_kernel(const _Float16* __restrict__ Vh,
                                   _Float16* __restrict__ Vt) {
  __shared__ __align__(16) _Float16 t[32][33];
  int bh = blockIdx.z, bq = bh >> 4, hh = bh & 15;
  int s0 = blockIdx.x * 32, d0 = blockIdx.y * 32;
  int tx = threadIdx.x, ty = threadIdx.y;
#pragma unroll
  for (int j = 0; j < 4; ++j)
    t[ty + 8 * j][tx] =
        Vh[(size_t)(bq * SEQ + s0 + ty + 8 * j) * D_MODEL + hh * HEAD_DIM + d0 + tx];
  __syncthreads();
#pragma unroll
  for (int j = 0; j < 4; ++j)
    Vt[(size_t)(bh * HEAD_DIM + d0 + ty + 8 * j) * SEQ + s0 + tx] = t[tx][ty + 8 * j];
}

// ------------------------------------------------------------------ WMMA GEMM
// C[M,N] = A[M,K] * Bt[N,K]^T   (A,Bt f16 row-major). 128x128 block tile,
// 8 waves (wm 0..3 x wn 0..1), each wave 32x64 via 2x4 WMMA tiles, K-step 64.
__global__ __launch_bounds__(256) void gemm_nt(
    const _Float16* __restrict__ A, const _Float16* __restrict__ Bt,
    float* __restrict__ Cf, _Float16* __restrict__ Ch,
    const float* __restrict__ bias, int M, int N, int K, int store_f32) {
  __shared__ __align__(16) _Float16 sA[128 * 64];
  __shared__ __align__(16) _Float16 sB[128 * 64];

  const int n0 = blockIdx.x * 128;
  const int m0 = blockIdx.y * 128;
  const int tid = threadIdx.x;
  const int wave = tid >> 5, lane = tid & 31;
  const int lo = lane & 15, half = lane >> 4;
  const int wm = wave >> 1, wn = wave & 1;

  v8f acc[2][4];
#pragma unroll
  for (int ms = 0; ms < 2; ++ms)
#pragma unroll
    for (int ns = 0; ns < 4; ++ns) { v8f z = {}; acc[ms][ns] = z; }

  for (int k0 = 0; k0 < K; k0 += 64) {
    __syncthreads();
    // 128 rows x 64 f16 per tile = 1024 16-byte chunks; 4 per thread per tile
    for (int i = tid; i < 1024; i += 256) {
      int row = i >> 3, q = i & 7;
      copy16_g2l(&sA[row * 64 + q * 8],
                 &A[(size_t)(m0 + row) * K + k0 + q * 8]);
      copy16_g2l(&sB[row * 64 + q * 8],
                 &Bt[(size_t)(n0 + row) * K + k0 + q * 8]);
    }
    async_fence();
    __syncthreads();

    HF af[2][2]; // [m-subtile][k-fragment], kept resident
#pragma unroll
    for (int ms = 0; ms < 2; ++ms) {
      const _Float16* ap = &sA[(wm * 32 + ms * 16 + lo) * 64];
#pragma unroll
      for (int kf = 0; kf < 2; ++kf) {
        af[ms][kf].h[0] = *(const v8h*)(ap + 32 * kf + 8 * half);
        af[ms][kf].h[1] = *(const v8h*)(ap + 32 * kf + 16 + 8 * half);
      }
    }
#pragma unroll
    for (int ns = 0; ns < 4; ++ns) {
      const _Float16* bp = &sB[(wn * 64 + ns * 16 + lo) * 64];
      HF bf[2];
#pragma unroll
      for (int kf = 0; kf < 2; ++kf) {
        bf[kf].h[0] = *(const v8h*)(bp + 32 * kf + 8 * half);
        bf[kf].h[1] = *(const v8h*)(bp + 32 * kf + 16 + 8 * half);
      }
#pragma unroll
      for (int ms = 0; ms < 2; ++ms) {
        acc[ms][ns] = wmma16(af[ms][0].v, bf[0].v, acc[ms][ns]);
        acc[ms][ns] = wmma16(af[ms][1].v, bf[1].v, acc[ms][ns]);
      }
    }
  }

#pragma unroll
  for (int ms = 0; ms < 2; ++ms)
#pragma unroll
    for (int ns = 0; ns < 4; ++ns)
#pragma unroll
      for (int r = 0; r < 8; ++r) {
        int m = m0 + wm * 32 + ms * 16 + r + 8 * half;
        int n = n0 + wn * 64 + ns * 16 + lo;
        float v = acc[ms][ns][r];
        if (store_f32)
          Cf[(size_t)m * N + n] = v + (bias ? bias[n] : 0.0f);
        else
          Ch[(size_t)m * N + n] = (_Float16)v;
      }
}

// ----------------------------------------------------------- flash attention
// One block = 64 queries of one (b,h); 4 waves x 16 query rows each.
// Streams causal key blocks of 64; online softmax; WMMA for QK^T and P*V.
__global__ __launch_bounds__(128) void attn_kernel(
    const _Float16* __restrict__ Qh, const _Float16* __restrict__ Kh,
    const _Float16* __restrict__ Vt, _Float16* __restrict__ ctxh) {
  __shared__ __align__(16) _Float16 sK[64 * 64];     // [t][d]
  __shared__ __align__(16) _Float16 sV[64 * 64];     // [d][t]
  __shared__ __align__(16) _Float16 sP[4 * 16 * 64]; // per wave [m][t]

  const int qblk = blockIdx.x;
  const int bh = blockIdx.y;
  const int b = bh >> 4, h = bh & 15;

  const int tid = threadIdx.x;
  const int wave = tid >> 5, lane = tid & 31;
  const int lo = lane & 15, half = lane >> 4;

  // Q fragments for this wave's 16 rows, pre-scaled by 1/sqrt(HEAD_DIM)=0.125
  HF qf[2];
  {
    const int qrow = b * SEQ + qblk * 64 + wave * 16 + lo;
    const _Float16* qp = Qh + (size_t)qrow * D_MODEL + h * HEAD_DIM;
#pragma unroll
    for (int f = 0; f < 2; ++f) {
      qf[f].h[0] = *(const v8h*)(qp + 32 * f + 8 * half);
      qf[f].h[1] = *(const v8h*)(qp + 32 * f + 16 + 8 * half);
#pragma unroll
      for (int e = 0; e < 16; ++e) qf[f].v[e] *= (_Float16)0.125f;
    }
  }

  v8f acc[4];
#pragma unroll
  for (int ds = 0; ds < 4; ++ds) { v8f z = {}; acc[ds] = z; }
  float m_run[8], l_run[8];
#pragma unroll
  for (int r = 0; r < 8; ++r) { m_run[r] = -1e30f; l_run[r] = 0.0f; }

  for (int kb = 0; kb <= qblk; ++kb) {
    __syncthreads();
    for (int i = tid; i < 512; i += 128) {
      int row = i >> 3, q = i & 7;
      copy16_g2l(&sK[row * 64 + q * 8],
                 &Kh[(size_t)(b * SEQ + kb * 64 + row) * D_MODEL +
                     h * HEAD_DIM + q * 8]);
      copy16_g2l(&sV[row * 64 + q * 8],
                 &Vt[(size_t)(bh * HEAD_DIM + row) * SEQ + kb * 64 + q * 8]);
    }
    async_fence();
    __syncthreads();

    // scores 16x64 = Q(16x64) * K^T
    v8f sc[4];
#pragma unroll
    for (int ns = 0; ns < 4; ++ns) {
      const _Float16* kp = &sK[(ns * 16 + lo) * 64];
      HF b0, b1;
      b0.h[0] = *(const v8h*)(kp + 8 * half);
      b0.h[1] = *(const v8h*)(kp + 16 + 8 * half);
      b1.h[0] = *(const v8h*)(kp + 32 + 8 * half);
      b1.h[1] = *(const v8h*)(kp + 32 + 16 + 8 * half);
      v8f c = {};
      c = wmma16(qf[0].v, b0.v, c);
      c = wmma16(qf[1].v, b1.v, c);
      sc[ns] = c;
    }

    if (kb == qblk) { // causal mask inside the diagonal block
#pragma unroll
      for (int ns = 0; ns < 4; ++ns)
#pragma unroll
        for (int r = 0; r < 8; ++r)
          if (ns * 16 + lo > wave * 16 + r + 8 * half) sc[ns][r] = -1e30f;
    }

    // online softmax (row stats reduced across 16-lane halves)
    float alpha[8];
#pragma unroll
    for (int r = 0; r < 8; ++r) {
      float mx = fmaxf(fmaxf(sc[0][r], sc[1][r]), fmaxf(sc[2][r], sc[3][r]));
#pragma unroll
      for (int off = 8; off; off >>= 1) mx = fmaxf(mx, __shfl_xor(mx, off, 32));
      float mn = fmaxf(m_run[r], mx);
      alpha[r] = __expf(m_run[r] - mn);
      m_run[r] = mn;
      float rs = 0.0f;
#pragma unroll
      for (int ns = 0; ns < 4; ++ns) {
        float p = __expf(sc[ns][r] - mn);
        rs += p;
        sP[wave * 1024 + (r + 8 * half) * 64 + ns * 16 + lo] = (_Float16)p;
      }
#pragma unroll
      for (int off = 8; off; off >>= 1) rs += __shfl_xor(rs, off, 32);
      l_run[r] = l_run[r] * alpha[r] + rs;
    }
#pragma unroll
    for (int ds = 0; ds < 4; ++ds)
#pragma unroll
      for (int r = 0; r < 8; ++r) acc[ds][r] *= alpha[r];

    asm volatile("s_wait_dscnt 0" ::: "memory"); // P stores visible to our loads

    // ctx += P(16x64) * V(64x64)
    HF pf[2];
    const _Float16* pp = &sP[wave * 1024 + lo * 64];
#pragma unroll
    for (int f = 0; f < 2; ++f) {
      pf[f].h[0] = *(const v8h*)(pp + 32 * f + 8 * half);
      pf[f].h[1] = *(const v8h*)(pp + 32 * f + 16 + 8 * half);
    }
#pragma unroll
    for (int ds = 0; ds < 4; ++ds) {
      const _Float16* vp = &sV[(ds * 16 + lo) * 64];
      HF b0, b1;
      b0.h[0] = *(const v8h*)(vp + 8 * half);
      b0.h[1] = *(const v8h*)(vp + 16 + 8 * half);
      b1.h[0] = *(const v8h*)(vp + 32 + 8 * half);
      b1.h[1] = *(const v8h*)(vp + 32 + 16 + 8 * half);
      acc[ds] = wmma16(pf[0].v, b0.v, acc[ds]);
      acc[ds] = wmma16(pf[1].v, b1.v, acc[ds]);
    }
  }

#pragma unroll
  for (int ds = 0; ds < 4; ++ds)
#pragma unroll
    for (int r = 0; r < 8; ++r) {
      float o = acc[ds][r] / l_run[r];
      int row = b * SEQ + qblk * 64 + wave * 16 + r + 8 * half;
      int col = h * HEAD_DIM + ds * 16 + lo;
      ctxh[(size_t)row * D_MODEL + col] = (_Float16)o;
    }
}

// ------------------------------------------------------------------- launcher

extern "C" void kernel_launch(void* const* d_in, const int* in_sizes, int n_in,
                              void* d_out, int out_size, void* d_ws,
                              size_t ws_size, hipStream_t stream) {
  (void)in_sizes; (void)n_in; (void)out_size; (void)ws_size;
  const float* x  = (const float*)d_in[0];
  const float* Wq = (const float*)d_in[1];
  const float* Wk = (const float*)d_in[2];
  const float* Wv = (const float*)d_in[3];
  const float* Wo = (const float*)d_in[4];
  const float* bo = (const float*)d_in[5];
  float* out = (float*)d_out;

  char* ws = (char*)d_ws;
  const size_t MB = 1024 * 1024;
  _Float16* xh  = (_Float16*)(ws + 0 * MB);   // 8 MB
  _Float16* WqT = (_Float16*)(ws + 8 * MB);   // 2 MB
  _Float16* WkT = (_Float16*)(ws + 10 * MB);
  _Float16* WvT = (_Float16*)(ws + 12 * MB);
  _Float16* WoT = (_Float16*)(ws + 14 * MB);
  _Float16* Qh  = (_Float16*)(ws + 16 * MB);  // 8 MB each
  _Float16* Kh  = (_Float16*)(ws + 24 * MB);
  _Float16* Vh  = (_Float16*)(ws + 32 * MB);
  _Float16* Vt  = (_Float16*)(ws + 40 * MB);
  _Float16* Ch  = (_Float16*)(ws + 48 * MB);  // total 56 MB

  int n = ROWS * D_MODEL;
  cvt_f32_to_f16<<<n / 256, 256, 0, stream>>>(x, xh, n);

  dim3 tb(32, 8);
  transpose_cvt_w<<<dim3(32, 32), tb, 0, stream>>>(Wq, WqT);
  transpose_cvt_w<<<dim3(32, 32), tb, 0, stream>>>(Wk, WkT);
  transpose_cvt_w<<<dim3(32, 32), tb, 0, stream>>>(Wv, WvT);
  transpose_cvt_w<<<dim3(32, 32), tb, 0, stream>>>(Wo, WoT);

  dim3 gg(D_MODEL / 128, ROWS / 128); // (8, 32)
  gemm_nt<<<gg, 256, 0, stream>>>(xh, WqT, nullptr, Qh, nullptr, ROWS, D_MODEL, D_MODEL, 0);
  gemm_nt<<<gg, 256, 0, stream>>>(xh, WkT, nullptr, Kh, nullptr, ROWS, D_MODEL, D_MODEL, 0);
  gemm_nt<<<gg, 256, 0, stream>>>(xh, WvT, nullptr, Vh, nullptr, ROWS, D_MODEL, D_MODEL, 0);

  transpose_v_kernel<<<dim3(SEQ / 32, HEAD_DIM / 32, BATCH * N_HEADS), tb, 0,
                       stream>>>(Vh, Vt);

  attn_kernel<<<dim3(SEQ / 64, BATCH * N_HEADS), 128, 0, stream>>>(Qh, Kh, Vt, Ch);

  gemm_nt<<<gg, 256, 0, stream>>>(Ch, WoT, out, nullptr, bo, ROWS, D_MODEL, D_MODEL, 1);
}
